// QuantizedLinear_7825430413556
// MI455X (gfx1250) — matrix operand order repaired and checked by reading, not compiled, not run on hardware
//
#include <hip/hip_runtime.h>

#define TOKENS  8192
#define IN_DIM  4096
#define OUT_DIM 4096
#define CB_V    8
#define RANK    128

typedef __attribute__((ext_vector_type(16))) __bf16 v16bf;
typedef __attribute__((ext_vector_type(8)))  float  v8f;

struct __attribute__((aligned(16))) U128 { unsigned int x, y, z, w; };
struct __attribute__((aligned(16))) F4   { float x, y, z, w; };
struct __attribute__((aligned(8)))  US4  { unsigned short x, y, z, w; };

__device__ __forceinline__ unsigned short f2bf(float f) {
  union { float f; unsigned u; } c; c.f = f;
  unsigned r = c.u + 0x7FFFu + ((c.u >> 16) & 1u);
  return (unsigned short)(r >> 16);
}

__device__ __forceinline__ void wait_async0() {
#if defined(__has_builtin)
#if __has_builtin(__builtin_amdgcn_s_wait_asynccnt)
  __builtin_amdgcn_s_wait_asynccnt(0);
#else
  asm volatile("s_wait_asynccnt 0x0" ::: "memory");
#endif
#else
  asm volatile("s_wait_asynccnt 0x0" ::: "memory");
#endif
}

// Async DMA of 16 bytes/lane straight into LDS (ASYNCcnt-tracked, no VGPR
// staging).  lds = wave-relative LDS byte address (low 32 bits of a generic
// shared pointer).
__device__ __forceinline__ void async_b128(unsigned lds, unsigned long long ga) {
  asm volatile("global_load_async_to_lds_b128 %0, %1, off"
               :: "v"(lds), "v"(ga) : "memory");
}

// ---------------------------------------------------------------------------
// Prep kernels (memory bound, trivial cost next to the 275-GFLOP GEMM)
// ---------------------------------------------------------------------------

// x (f32) -> Xb (bf16), plus per-token row sum for the wbias term.
__global__ __launch_bounds__(256) void qlin_prep_x(
    const float* __restrict__ x, unsigned short* __restrict__ Xb,
    float* __restrict__ xsum) {
  const int t = blockIdx.x;
  const F4* row = (const F4*)(x + (size_t)t * IN_DIM);
  US4* orow = (US4*)(Xb + (size_t)t * IN_DIM);
  float s = 0.f;
#pragma unroll
  for (int j = 0; j < 4; ++j) {
    const int i = threadIdx.x + j * 256;          // 0..1023 float4s
    F4 v = row[i];
    s += v.x + v.y + v.z + v.w;
    US4 o; o.x = f2bf(v.x); o.y = f2bf(v.y); o.z = f2bf(v.z); o.w = f2bf(v.w);
    orow[i] = o;
  }
  __shared__ float red[256];
  red[threadIdx.x] = s;
  __syncthreads();
  for (int off = 128; off > 0; off >>= 1) {
    if (threadIdx.x < off) red[threadIdx.x] += red[threadIdx.x + off];
    __syncthreads();
  }
  if (threadIdx.x == 0) xsum[t] = red[0];
}

// Codebook dequant: Wb[o,i] = centroids[idx[o][i/8]][i%8] * scale[o]  (bf16)
__global__ __launch_bounds__(256) void qlin_prep_w(
    const float* __restrict__ cent, const int* __restrict__ indices,
    const float* __restrict__ wscale, unsigned short* __restrict__ Wb) {
  const int o = blockIdx.x;
  const float sc = wscale[o];
  const int* ir = indices + (size_t)o * (IN_DIM / CB_V);
  U128* wrow = (U128*)(Wb + (size_t)o * IN_DIM);
  for (int g = threadIdx.x; g < IN_DIM / CB_V; g += 256) {
    const int c = ir[g];
    const float* cen = cent + (size_t)c * CB_V;
    unsigned short b[8];
#pragma unroll
    for (int k = 0; k < 8; ++k) b[k] = f2bf(cen[k] * sc);
    U128 pk;
    pk.x = (unsigned)b[0] | ((unsigned)b[1] << 16);
    pk.y = (unsigned)b[2] | ((unsigned)b[3] << 16);
    pk.z = (unsigned)b[4] | ((unsigned)b[5] << 16);
    pk.w = (unsigned)b[6] | ((unsigned)b[7] << 16);
    wrow[g] = pk;
  }
}

// VtS[r,i] = S[r] * Vt[r,i]  (bf16)
__global__ __launch_bounds__(256) void qlin_prep_vts(
    const float* __restrict__ Vt, const float* __restrict__ S,
    unsigned short* __restrict__ VtS) {
  const int r = blockIdx.x;
  const float s = S[r];
  const F4* row = (const F4*)(Vt + (size_t)r * IN_DIM);
  US4* orow = (US4*)(VtS + (size_t)r * IN_DIM);
  for (int i = threadIdx.x; i < IN_DIM / 4; i += 256) {
    F4 v = row[i];
    US4 o; o.x = f2bf(v.x * s); o.y = f2bf(v.y * s);
    o.z = f2bf(v.z * s); o.w = f2bf(v.w * s);
    orow[i] = o;
  }
}

// U (f32) -> Ub (bf16)
__global__ __launch_bounds__(256) void qlin_prep_u(
    const float* __restrict__ U, unsigned short* __restrict__ Ub) {
  const size_t i = (size_t)blockIdx.x * 256 + threadIdx.x;  // float4 index
  F4 v = ((const F4*)U)[i];
  US4 o; o.x = f2bf(v.x); o.y = f2bf(v.y); o.z = f2bf(v.z); o.w = f2bf(v.w);
  ((US4*)Ub)[i] = o;
}

// ---------------------------------------------------------------------------
// bf16 WMMA GEMM:  C[M,N] = A[M,K1] @ B1[N,K1]^T  (+ A2[M,K2] @ B2[N,K2]^T)
// Block tile 128x128, 8 waves (wave32) in a 4(M) x 2(N) grid, each wave owns
// 32x64 = 8 accumulator tiles of 16x16.  K-step 32, double-buffered LDS
// filled by GLOBAL_LOAD_ASYNC_TO_LDS_B128 (ASYNCcnt), padded row stride
// (40 bf16) for conflict-free b128 fragment reads.
// ---------------------------------------------------------------------------
#define LDS_STRIDE 40  // 32 K elems + 8 pad (bf16); 80 bytes (16B multiple)

template <bool MAIN>
__global__ __launch_bounds__(256) void qlin_gemm(
    const unsigned short* __restrict__ A1, int lda1, int k1,
    const unsigned short* __restrict__ B1, int ldb1,
    const unsigned short* __restrict__ A2, int lda2, int k2,
    const unsigned short* __restrict__ B2, int ldb2,
    const float* __restrict__ bias, const float* __restrict__ wbias,
    const float* __restrict__ xsum,
    float* __restrict__ outF, unsigned short* __restrict__ outB, int ldc) {
  __shared__ unsigned short As[2][128 * LDS_STRIDE];
  __shared__ unsigned short Bs[2][128 * LDS_STRIDE];

  const int tid  = threadIdx.x;
  const int lane = tid & 31;
  const int wid  = tid >> 5;
  const int wm   = wid & 3;   // 0..3  -> M offset wm*32
  const int wn   = wid >> 2;  // 0..1  -> N offset wn*64
  const int m0   = blockIdx.y * 128;
  const int n0   = blockIdx.x * 128;

  const int rI = tid >> 2;  // 0..63 : tile row loaded by this thread
  const int c4 = tid & 3;   // 0..3  : 8-element K chunk

  const int s1     = k1 >> 5;
  const int nsteps = s1 + (k2 >> 5);

  // Per-thread LDS destinations (wave-relative byte addresses).
  const unsigned ldsA0 =
      (unsigned)(uintptr_t)((char*)&As[0][0] + (size_t)rI * 80 + c4 * 16);
  const unsigned ldsB0 =
      (unsigned)(uintptr_t)((char*)&Bs[0][0] + (size_t)rI * 80 + c4 * 16);
  const unsigned bufStride = 128 * LDS_STRIDE * 2;  // bytes per buffer
  const unsigned rowHalf   = 64 * 80;               // rows 64..127

  auto issue_async = [&](int s, int buf) {
    const unsigned short *Ab, *Bb; int la, lb, kb;
    if (s < s1) { Ab = A1; Bb = B1; la = lda1; lb = ldb1; kb = s << 5; }
    else        { Ab = A2; Bb = B2; la = lda2; lb = ldb2; kb = (s - s1) << 5; }
    const int ko = kb + c4 * 8;
    const unsigned long long ga0 =
        (unsigned long long)(uintptr_t)(Ab + (size_t)(m0 + rI) * la + ko);
    const unsigned long long ga1 =
        (unsigned long long)(uintptr_t)(Ab + (size_t)(m0 + rI + 64) * la + ko);
    const unsigned long long gb0 =
        (unsigned long long)(uintptr_t)(Bb + (size_t)(n0 + rI) * lb + ko);
    const unsigned long long gb1 =
        (unsigned long long)(uintptr_t)(Bb + (size_t)(n0 + rI + 64) * lb + ko);
    const unsigned la0 = ldsA0 + (unsigned)buf * bufStride;
    const unsigned lb0 = ldsB0 + (unsigned)buf * bufStride;
    async_b128(la0,           ga0);
    async_b128(la0 + rowHalf, ga1);
    async_b128(lb0,           gb0);
    async_b128(lb0 + rowHalf, gb1);
  };

  v8f acc[2][4];
  const v8f vzero = {0.f, 0.f, 0.f, 0.f, 0.f, 0.f, 0.f, 0.f};
#pragma unroll
  for (int i = 0; i < 2; ++i)
#pragma unroll
    for (int j = 0; j < 4; ++j) acc[i][j] = vzero;

  // ISA 7.12.2 lane layouts (16-bit A 16x32 / B 32x16), byte offsets in a row
  const int halfSel = (lane < 16) ? 0 : 1;
  const int rowIdx  = lane & 15;
  const int offA0   = halfSel ? 16 : 0;  // A: K {0..7,16..23} vs {8..15,24..31}
  const int offB0   = halfSel ? 16 : 0;  // B: K {0..7,16..23} vs {8..15,24..31}

  issue_async(0, 0);
  wait_async0();
  __syncthreads();

  for (int s = 0; s < nsteps; ++s) {
    const int cur = s & 1;
    const bool more = (s + 1) < nsteps;
    if (more) issue_async(s + 1, cur ^ 1);
    if (s + 2 < nsteps) {  // L2 prefetch of tile k+2 -> global_prefetch_b8
      const int s2 = s + 2;
      const unsigned short *Ab, *Bb; int la, lb, kb;
      if (s2 < s1) { Ab = A1; Bb = B1; la = lda1; lb = ldb1; kb = s2 << 5; }
      else         { Ab = A2; Bb = B2; la = lda2; lb = ldb2; kb = (s2 - s1) << 5; }
      __builtin_prefetch(Ab + (size_t)(m0 + rI) * la + kb + c4 * 8, 0, 0);
      __builtin_prefetch(Bb + (size_t)(n0 + rI) * lb + kb + c4 * 8, 0, 0);
    }

    union Frag { v16bf v; U128 q[2]; };
    Frag a[2], b[4];
#pragma unroll
    for (int tm = 0; tm < 2; ++tm) {
      const char* p = (const char*)&As[cur][0] +
                      (size_t)(wm * 32 + tm * 16 + rowIdx) * 80 + offA0;
      a[tm].q[0] = *(const U128*)p;
      a[tm].q[1] = *(const U128*)(p + 32);
    }
#pragma unroll
    for (int tn = 0; tn < 4; ++tn) {
      const char* p = (const char*)&Bs[cur][0] +
                      (size_t)(wn * 64 + tn * 16 + rowIdx) * 80 + offB0;
      b[tn].q[0] = *(const U128*)p;
      b[tn].q[1] = *(const U128*)(p + 32);
    }
#pragma unroll
    for (int tm = 0; tm < 2; ++tm)
#pragma unroll
      for (int tn = 0; tn < 4; ++tn)
        acc[tm][tn] = __builtin_amdgcn_wmma_f32_16x16x32_bf16(
            false, a[tm].v, false, b[tn].v, (short)0, acc[tm][tn], false, false);

    if (more) wait_async0();  // next buffer fully in LDS before the barrier
    __syncthreads();
  }

  // Epilogue.  C tile layout (ISA 7.12.2): lane -> col = lane&15,
  // vgpr g -> row = g + (lane<16 ? 0 : 8).
#pragma unroll
  for (int tm = 0; tm < 2; ++tm) {
#pragma unroll
    for (int tn = 0; tn < 4; ++tn) {
      const int mb = m0 + wm * 32 + tm * 16 + (halfSel ? 8 : 0);
      const int n  = n0 + wn * 64 + tn * 16 + rowIdx;
      if (MAIN) {
        const float bv = bias[n];
        const float wv = wbias[n];
#pragma unroll
        for (int g = 0; g < 8; ++g) {
          const int m = mb + g;
          outF[(size_t)m * ldc + n] = acc[tm][tn][g] + bv + wv * xsum[m];
        }
      } else {
#pragma unroll
        for (int g = 0; g < 8; ++g) {
          const int m = mb + g;
          outB[(size_t)m * ldc + n] = f2bf(acc[tm][tn][g]);
        }
      }
    }
  }
}

// ---------------------------------------------------------------------------
// Launch
// ---------------------------------------------------------------------------
extern "C" void kernel_launch(void* const* d_in, const int* in_sizes, int n_in,
                              void* d_out, int out_size, void* d_ws,
                              size_t ws_size, hipStream_t stream) {
  const float* x       = (const float*)d_in[0];
  const float* cent    = (const float*)d_in[1];
  const int*   indices = (const int*)d_in[2];
  const float* U       = (const float*)d_in[3];
  const float* S       = (const float*)d_in[4];
  const float* Vt      = (const float*)d_in[5];
  const float* wscale  = (const float*)d_in[6];
  const float* wbias   = (const float*)d_in[7];
  const float* bias    = (const float*)d_in[8];
  float* out = (float*)d_out;

  // workspace layout (bytes)
  char* ws = (char*)d_ws;
  unsigned short* Xb  = (unsigned short*)(ws);                         // 67108864
  unsigned short* Wb  = (unsigned short*)(ws + 67108864);              // 33554432
  unsigned short* T1b = (unsigned short*)(ws + 100663296);             //  2097152
  unsigned short* VtS = (unsigned short*)(ws + 102760448);             //  1048576
  unsigned short* Ub  = (unsigned short*)(ws + 103809024);             //  1048576
  float*          xsv = (float*)(ws + 104857600);                      //    32768

  qlin_prep_x  <<<TOKENS,  256, 0, stream>>>(x, Xb, xsv);
  qlin_prep_w  <<<OUT_DIM, 256, 0, stream>>>(cent, indices, wscale, Wb);
  qlin_prep_vts<<<RANK,    256, 0, stream>>>(Vt, S, VtS);
  qlin_prep_u  <<<(OUT_DIM * RANK) / (256 * 4), 256, 0, stream>>>(U, Ub);

  // T1 = Xb @ VtS^T   (M=TOKENS, N=RANK, K=IN_DIM), bf16 result
  qlin_gemm<false><<<dim3(1, TOKENS / 128), 256, 0, stream>>>(
      Xb, IN_DIM, IN_DIM, VtS, IN_DIM,
      nullptr, 0, 0, nullptr, 0,
      nullptr, nullptr, nullptr, nullptr, T1b, RANK);

  // out = Xb @ Wb^T + T1 @ Ub^T + bias + wbias * xsum   (M=TOKENS, N=OUT_DIM)
  qlin_gemm<true><<<dim3(OUT_DIM / 128, TOKENS / 128), 256, 0, stream>>>(
      Xb, IN_DIM, IN_DIM, Wb, IN_DIM,
      T1b, RANK, RANK, Ub, RANK,
      bias, wbias, xsv, out, nullptr, OUT_DIM);
}
